// OneTransBlock_9500467659303
// MI455X (gfx1250) — matrix operand order repaired and checked by reading, not compiled, not run on hardware
//
#include <hip/hip_runtime.h>

// ---------------- constants (match reference) ----------------
constexpr int BB   = 4;
constexpr int T    = 2048;
constexpr int D    = 768;
constexpr int H    = 12;
constexpr int NS   = 32;
constexpr int FFN  = 3072;
constexpr int DH   = 64;        // D / H
constexpr int NSTOK = T - NS;   // 2016 shared-weight tokens
constexpr int QKVW = 3 * D;     // 2304

typedef __attribute__((ext_vector_type(16))) __bf16 v16bf;
typedef __attribute__((ext_vector_type(8)))  float  v8f;
typedef __attribute__((ext_vector_type(4)))  unsigned int v4u;

// ---------------- helpers ----------------
static __device__ inline unsigned short bf16bits(float x) {
  unsigned int u = __builtin_bit_cast(unsigned int, x);
  u += 0x7FFFu + ((u >> 16) & 1u);            // round-to-nearest-even
  return (unsigned short)(u >> 16);
}
static __device__ inline float bf2f(unsigned short s) {
  return __builtin_bit_cast(float, ((unsigned int)s) << 16);
}
static __device__ inline float gelu_exact(float x) {
  return 0.5f * x * (1.0f + erff(x * 0.70710678118654752f));
}
static __device__ inline v8f wmma_bf16(v16bf a, v16bf b, v8f c) {
  return __builtin_amdgcn_wmma_f32_16x16x32_bf16(false, a, false, b, (short)0, c,
                                                 false, false);
}

// Assemble a 16-element bf16 fragment from two 16-byte chunks (no VALU work).
static __device__ inline v16bf frag_from2(const unsigned short* p0,
                                          const unsigned short* p1) {
  v4u lo = *(const v4u*)p0;
  v4u hi = *(const v4u*)p1;
  return __builtin_bit_cast(v16bf,
      __builtin_shufflevector(lo, hi, 0, 1, 2, 3, 4, 5, 6, 7));
}

// A fragment (16M x 32K), row-major bf16; rowbase -> (mlane, khalf)
// per-lane K layout: VGPR r<4 -> K=2r+khalf ; r>=4 -> K=16+2(r-4)+khalf
static __device__ inline v16bf load_a_frag(const unsigned short* rowbase) {
  return frag_from2(rowbase, rowbase + 16);
}

// B fragment (32K x 16N), per-lane: K = kg+0..7 and kg+16..23 (same order as A)
static __device__ inline v16bf load_b_frag(const unsigned short* rowbase) {
  return frag_from2(rowbase, rowbase + 16);
}

// ---------------- f32 -> bf16 bulk convert (weights, once) ----------------
__global__ __launch_bounds__(256) void cvt_bf16_kernel(const float* __restrict__ in,
                                                       unsigned short* __restrict__ out,
                                                       long n) {
  long i = (long)blockIdx.x * 256 + threadIdx.x;
  long stride = (long)gridDim.x * 256;
  for (; i < n; i += stride) out[i] = bf16bits(in[i]);
}

// ---------------- RMSNorm (f32 in -> bf16 bits out) ----------------
__global__ __launch_bounds__(256) void rmsnorm_kernel(const float* __restrict__ x,
                                                      const float* __restrict__ w,
                                                      unsigned short* __restrict__ out) {
  long row = blockIdx.x;
  const float* xr = x + row * D;
  float ss = 0.0f;
#pragma unroll
  for (int i = threadIdx.x; i < D; i += 256) {
    float v = xr[i];
    ss += v * v;
  }
  for (int i = 1; i < 32; i <<= 1) ss += __shfl_xor(ss, i);
  __shared__ float wsum[8];
  if ((threadIdx.x & 31) == 0) wsum[threadIdx.x >> 5] = ss;
  __syncthreads();
  float tot = 0.0f;
#pragma unroll
  for (int i = 0; i < 8; ++i) tot += wsum[i];
  float rms = rsqrtf(tot / (float)D + 1e-6f);
#pragma unroll
  for (int i = threadIdx.x; i < D; i += 256)
    out[row * D + i] = bf16bits(xr[i] * rms * w[i]);
}

// ---------------- V transpose: qkv16 V-part -> vT[b][h][dh][t] (bf16) ----------------
// grid = (B*H, T/64), block = 256; LDS-tiled 64 tokens x 64 dh
__global__ __launch_bounds__(256) void transpose_v_kernel(
    const unsigned short* __restrict__ qkv, unsigned short* __restrict__ vT) {
  __shared__ unsigned short tile[64][68];   // pad to stagger banks
  int b = blockIdx.x / H, h = blockIdx.x % H;
  int t0 = blockIdx.y * 64;
#pragma unroll
  for (int c = threadIdx.x; c < 64 * 16; c += 256) {
    int row = c >> 4;           // token in tile
    int col = (c & 15) * 4;     // dh
    const unsigned short* src =
        qkv + ((long)(b * T + t0 + row)) * QKVW + 2 * D + h * DH + col;
    unsigned long long v = *(const unsigned long long*)src;
    tile[row][col]     = (unsigned short)v;
    tile[row][col + 1] = (unsigned short)(v >> 16);
    tile[row][col + 2] = (unsigned short)(v >> 32);
    tile[row][col + 3] = (unsigned short)(v >> 48);
  }
  __syncthreads();
#pragma unroll
  for (int c = threadIdx.x; c < 64 * 16; c += 256) {
    int dh = c >> 4;
    int tc = (c & 15) * 4;      // token in tile
    unsigned long long v =
        (unsigned long long)tile[tc][dh] |
        ((unsigned long long)tile[tc + 1][dh] << 16) |
        ((unsigned long long)tile[tc + 2][dh] << 32) |
        ((unsigned long long)tile[tc + 3][dh] << 48);
    *(unsigned long long*)(vT + ((long)(b * H + h) * DH + dh) * T + t0 + tc) = v;
  }
}

// ---------------- WMMA GEMM: C = A(bf16, MxK) * W16(bf16, NxK)^T ----------------
// block = 256 (8 waves, 2x4), block tile 64M x 128N, wave tile 32M x 32N
template <bool GELU, bool OUT_BF16>
__global__ __launch_bounds__(256) void gemm_wmma_kernel(
    const unsigned short* __restrict__ A, long strideAb,
    const unsigned short* __restrict__ W16,
    float* __restrict__ Cf, unsigned short* __restrict__ Cb, long strideCb,
    const float* __restrict__ resid, long strideRb,
    int M, int N, int K) {
  int bz = blockIdx.z;
  A += (long)bz * strideAb;
  int wave = threadIdx.x >> 5, lane = threadIdx.x & 31;
  int half = lane >> 4, nl = lane & 15;
  int m0 = blockIdx.x * 64 + (wave >> 2) * 32;
  int n0 = blockIdx.y * 128 + (wave & 3) * 32;

  int mr0 = m0 + nl;        if (mr0 >= M) mr0 = M - 1;   // clamp; store guard discards
  int mr1 = m0 + 16 + nl;   if (mr1 >= M) mr1 = M - 1;
  const unsigned short* a0 = A + (long)mr0 * K + half * 8;
  const unsigned short* a1 = A + (long)mr1 * K + half * 8;
  const unsigned short* b0 = W16 + (long)(n0 + nl) * K + half * 8;
  const unsigned short* b1 = b0 + 16L * K;

  v8f c00 = {}, c01 = {}, c10 = {}, c11 = {};
  for (int k0 = 0; k0 < K; k0 += 32) {
    v16bf af0 = load_a_frag(a0 + k0);
    v16bf af1 = load_a_frag(a1 + k0);
    v16bf bf0 = load_b_frag(b0 + k0);
    v16bf bf1 = load_b_frag(b1 + k0);
    c00 = wmma_bf16(af0, bf0, c00);
    c01 = wmma_bf16(af0, bf1, c01);
    c10 = wmma_bf16(af1, bf0, c10);
    c11 = wmma_bf16(af1, bf1, c11);
  }

#pragma unroll
  for (int r = 0; r < 8; ++r) {
#pragma unroll
    for (int sm = 0; sm < 2; ++sm) {
      int m = m0 + sm * 16 + r + 8 * half;
      if (m < M) {
        float v0 = sm ? c10[r] : c00[r];
        float v1 = sm ? c11[r] : c01[r];
        long idx = (long)m * N + n0 + nl;
        if constexpr (GELU) {
          v0 = gelu_exact(v0);
          v1 = gelu_exact(v1);
        }
        if (resid) {
          v0 += resid[(long)bz * strideRb + idx];
          v1 += resid[(long)bz * strideRb + idx + 16];
        }
        if constexpr (OUT_BF16) {
          Cb[(long)bz * strideCb + idx]      = bf16bits(v0);
          Cb[(long)bz * strideCb + idx + 16] = bf16bits(v1);
        } else {
          Cf[(long)bz * strideCb + idx]      = v0;
          Cf[(long)bz * strideCb + idx + 16] = v1;
        }
      }
    }
  }
}

// ---------------- Flash attention (causal), WMMA bf16 ----------------
// grid = (B*H, T/128), block = 256 (8 waves x 16-query tiles)
__global__ __launch_bounds__(256) void attn_kernel(const unsigned short* __restrict__ qkv,
                                                   const unsigned short* __restrict__ vT,
                                                   unsigned short* __restrict__ attn_out) {
  __shared__ __align__(16) unsigned short pshare[8][16 * 32];
  int b = blockIdx.x / H, h = blockIdx.x % H;
  int wave = threadIdx.x >> 5, lane = threadIdx.x & 31;
  int half = lane >> 4, nl = lane & 15;
  int q0 = blockIdx.y * 128 + wave * 16;

  const float scale = 0.125f;  // 1/sqrt(DH), applied post-WMMA
  const unsigned short* qbase =
      qkv + ((long)(b * T + q0 + nl)) * QKVW + h * DH + half * 8;
  v16bf aq0 = load_a_frag(qbase);        // dh 0..31
  v16bf aq1 = load_a_frag(qbase + 32);   // dh 32..63

  v8f acc0 = {}, acc1 = {}, acc2 = {}, acc3 = {};
  float rmax[8], rsum[8];
#pragma unroll
  for (int r = 0; r < 8; ++r) { rmax[r] = -INFINITY; rsum[r] = 0.0f; }

  unsigned short* ps = &pshare[wave][0];
  // vT rows for this head: row dh, contiguous over t, ld = T
  const unsigned short* vhead = vT + ((long)(b * H + h) * DH) * T;
  int jend = (q0 + 15) >> 5;
  for (int j = 0; j <= jend; ++j) {
    int key0 = j * 32;
    // K fragments: rows = keys, contiguous dh -> pure vector loads
    const unsigned short* kbase =
        qkv + ((long)(b * T + key0 + nl)) * QKVW + D + h * DH + half * 8;
    v8f s0 = {}, s1 = {};
    s0 = wmma_bf16(aq0, load_b_frag(kbase), s0);
    s0 = wmma_bf16(aq1, load_b_frag(kbase + 32), s0);
    s1 = wmma_bf16(aq0, load_b_frag(kbase + 16L * QKVW), s1);
    s1 = wmma_bf16(aq1, load_b_frag(kbase + 16L * QKVW + 32), s1);

#pragma unroll
    for (int r = 0; r < 8; ++r) {
      int q = q0 + r + 8 * half;
      float v0 = (key0 + nl <= q) ? s0[r] * scale : -3.0e38f;
      float v1 = (key0 + nl + 16 <= q) ? s1[r] * scale : -3.0e38f;
      float mloc = fmaxf(v0, v1);
      for (int i = 1; i < 16; i <<= 1) mloc = fmaxf(mloc, __shfl_xor(mloc, i));
      float mnew = fmaxf(rmax[r], mloc);
      float corr = __expf(rmax[r] - mnew);
      rmax[r] = mnew;
      float e0 = __expf(v0 - mnew);
      float e1 = __expf(v1 - mnew);
      float sl = e0 + e1;
      for (int i = 1; i < 16; i <<= 1) sl += __shfl_xor(sl, i);
      rsum[r] = rsum[r] * corr + sl;
      acc0[r] *= corr; acc1[r] *= corr; acc2[r] *= corr; acc3[r] *= corr;
      int m = r + 8 * half;
      ps[m * 32 + nl]      = bf16bits(e0);
      ps[m * 32 + nl + 16] = bf16bits(e1);
    }
    asm volatile("s_wait_dscnt 0" ::: "memory");  // P tile visible within wave
    v16bf pf = load_a_frag(ps + (long)nl * 32 + half * 8);
    // V fragments from vT: K(=key)-contiguous -> pure vector loads
    const unsigned short* vtb = vhead + key0 + half * 8;
    acc0 = wmma_bf16(pf, load_b_frag(vtb + (long)(nl)      * T), acc0);
    acc1 = wmma_bf16(pf, load_b_frag(vtb + (long)(16 + nl) * T), acc1);
    acc2 = wmma_bf16(pf, load_b_frag(vtb + (long)(32 + nl) * T), acc2);
    acc3 = wmma_bf16(pf, load_b_frag(vtb + (long)(48 + nl) * T), acc3);
  }

#pragma unroll
  for (int r = 0; r < 8; ++r) {
    int q = q0 + r + 8 * half;
    float inv = 1.0f / rsum[r];
    long base = ((long)(b * T + q)) * D + h * DH + nl;
    attn_out[base + 0]  = bf16bits(acc0[r] * inv);
    attn_out[base + 16] = bf16bits(acc1[r] * inv);
    attn_out[base + 32] = bf16bits(acc2[r] * inv);
    attn_out[base + 48] = bf16bits(acc3[r] * inv);
  }
}

// ---------------- per-token (ns) GEMV, 4-batch weight reuse ----------------
// grid = (R/8, NS); each wave: one output row for all 4 batches
__global__ __launch_bounds__(256) void gemv_ns_kernel(
    const float* __restrict__ Wall,  // (NS, R, K) f32 (streamed once from HBM)
    const unsigned short* __restrict__ act, long actBatch, long actTokStride,
    int actTokBase,
    float* __restrict__ outF, unsigned short* __restrict__ outB,
    long outBatch, long outTokStride, int outTokBase,
    const float* __restrict__ resid,
    int R, int K, int gelu_flag) {
  int t = blockIdx.y;
  int wave = threadIdx.x >> 5, lane = threadIdx.x & 31;
  int row = blockIdx.x * 8 + wave;
  const float* wrow = Wall + ((long)t * R + row) * K;
  float acc[BB] = {0.f, 0.f, 0.f, 0.f};
  for (int k = lane; k < K; k += 32) {
    __builtin_prefetch(wrow + k + 256, 0, 1);  // stream next weight chunk
    float w = wrow[k];
#pragma unroll
    for (int bb = 0; bb < BB; ++bb)
      acc[bb] += w * bf2f(act[(long)bb * actBatch +
                              (long)(actTokBase + t) * actTokStride + k]);
  }
#pragma unroll
  for (int bb = 0; bb < BB; ++bb)
    for (int i = 1; i < 32; i <<= 1) acc[bb] += __shfl_xor(acc[bb], i);
  if (lane == 0) {
#pragma unroll
    for (int bb = 0; bb < BB; ++bb) {
      float v = acc[bb];
      if (gelu_flag) v = gelu_exact(v);
      long oidx = (long)bb * outBatch + (long)(outTokBase + t) * outTokStride + row;
      if (resid) v += resid[oidx];
      if (outB) outB[oidx] = bf16bits(v);
      else      outF[oidx] = v;
    }
  }
}

// ---------------- host launcher ----------------
extern "C" void kernel_launch(void* const* d_in, const int* in_sizes, int n_in,
                              void* d_out, int out_size, void* d_ws, size_t ws_size,
                              hipStream_t stream) {
  (void)in_sizes; (void)n_in; (void)out_size; (void)ws_size;
  const float* x          = (const float*)d_in[0];
  /* d_in[1] mask: causal, applied analytically */
  const float* s_qkv_w    = (const float*)d_in[2];
  const float* ns_qkv_w   = (const float*)d_in[3];
  const float* out_proj_w = (const float*)d_in[4];
  const float* s_w1       = (const float*)d_in[5];
  const float* s_w2       = (const float*)d_in[6];
  const float* ns_w1      = (const float*)d_in[7];
  const float* ns_w2      = (const float*)d_in[8];
  const float* norm1_w    = (const float*)d_in[9];
  const float* norm2_w    = (const float*)d_in[10];
  float* out = (float*)d_out;

  char* ws = (char*)d_ws;
  auto takeU16 = [&](size_t n) { unsigned short* p = (unsigned short*)ws; ws += n * 2; return p; };
  auto takeF32 = [&](size_t n) { float* p = (float*)ws; ws += n * 4; return p; };
  unsigned short* normed1 = takeU16((size_t)BB * T * D);
  unsigned short* qkv16   = takeU16((size_t)BB * T * QKVW);
  unsigned short* vT      = takeU16((size_t)BB * H * DH * T);
  unsigned short* attn_o  = takeU16((size_t)BB * T * D);
  float*          z       = takeF32((size_t)BB * T * D);
  unsigned short* normed2 = takeU16((size_t)BB * T * D);
  unsigned short* h_s     = takeU16((size_t)BB * NSTOK * FFN);
  unsigned short* h_ns    = takeU16((size_t)BB * NS * FFN);
  unsigned short* wqkv16  = takeU16((size_t)QKVW * D);
  unsigned short* wo16    = takeU16((size_t)D * D);
  unsigned short* w1_16   = takeU16((size_t)FFN * D);
  unsigned short* w2_16   = takeU16((size_t)D * FFN);

  // 0) one-time bf16 conversion of the shared (L2-resident) weights
  cvt_bf16_kernel<<<4096, 256, 0, stream>>>(s_qkv_w, wqkv16, (long)QKVW * D);
  cvt_bf16_kernel<<<2048, 256, 0, stream>>>(out_proj_w, wo16, (long)D * D);
  cvt_bf16_kernel<<<4096, 256, 0, stream>>>(s_w1, w1_16, (long)FFN * D);
  cvt_bf16_kernel<<<4096, 256, 0, stream>>>(s_w2, w2_16, (long)D * FFN);

  // 1) RMSNorm1
  rmsnorm_kernel<<<BB * T, 256, 0, stream>>>(x, norm1_w, normed1);

  // 2) shared QKV: (2016 x 768) x (2304 x 768)^T per batch -> bf16 qkv
  gemm_wmma_kernel<false, true><<<dim3((NSTOK + 63) / 64, QKVW / 128, BB), 256, 0, stream>>>(
      normed1, (long)T * D, wqkv16, nullptr, qkv16, (long)T * QKVW,
      nullptr, 0, NSTOK, QKVW, D);

  // 3) per-token QKV (last 32 tokens) -> bf16 qkv
  gemv_ns_kernel<<<dim3(QKVW / 8, NS), 256, 0, stream>>>(
      ns_qkv_w, normed1, (long)T * D, D, NSTOK,
      nullptr, qkv16, (long)T * QKVW, QKVW, NSTOK, nullptr, QKVW, D, 0);

  // 3b) transpose V for K-contiguous PV fragments
  transpose_v_kernel<<<dim3(BB * H, T / 64), 256, 0, stream>>>(qkv16, vT);

  // 4) causal flash attention
  attn_kernel<<<dim3(BB * H, T / 128), 256, 0, stream>>>(qkv16, vT, attn_o);

  // 5) out projection + residual: z = attn @ Wo^T + x   (M = B*T contiguous)
  gemm_wmma_kernel<false, false><<<dim3((BB * T) / 64, D / 128, 1), 256, 0, stream>>>(
      attn_o, 0, wo16, z, nullptr, 0, x, 0, BB * T, D, D);

  // 6) RMSNorm2
  rmsnorm_kernel<<<BB * T, 256, 0, stream>>>(z, norm2_w, normed2);

  // 7) shared FFN up + gelu -> h_s (bf16)
  gemm_wmma_kernel<true, true><<<dim3((NSTOK + 63) / 64, FFN / 128, BB), 256, 0, stream>>>(
      normed2, (long)T * D, w1_16, nullptr, h_s, (long)NSTOK * FFN,
      nullptr, 0, NSTOK, FFN, D);

  // 8) per-token FFN up + gelu -> h_ns (bf16)
  gemv_ns_kernel<<<dim3(FFN / 8, NS), 256, 0, stream>>>(
      ns_w1, normed2, (long)T * D, D, NSTOK,
      nullptr, h_ns, (long)NS * FFN, FFN, 0, nullptr, FFN, D, 1);

  // 9) shared FFN down + residual -> out (first 2016 tokens per batch)
  gemm_wmma_kernel<false, false><<<dim3((NSTOK + 63) / 64, D / 128, BB), 256, 0, stream>>>(
      h_s, (long)NSTOK * FFN, w2_16, out, nullptr, (long)T * D,
      z, (long)T * D, NSTOK, D, FFN);

  // 10) per-token FFN down + residual -> out (last 32 tokens per batch)
  gemv_ns_kernel<<<dim3(D / 8, NS), 256, 0, stream>>>(
      ns_w2, h_ns, (long)NS * FFN, FFN, 0,
      out, nullptr, (long)T * D, D, NSTOK, z, D, FFN, 0);
}
